// MultiHeadedAttention_36893769073274
// MI455X (gfx1250) — matrix operand order repaired and checked by reading, not compiled, not run on hardware
//
#include <hip/hip_runtime.h>
#include <hip/hip_bf16.h>
#include <math.h>

// Problem constants
#define BB 8
#define LL 1024
#define DD 1024
#define HH_HEADS 16
#define DPH 64
#define NREL 65   // 2*MAXREL+1

typedef __attribute__((ext_vector_type(16))) __bf16 v16bf;
typedef __attribute__((ext_vector_type(8)))  __bf16 v8bf;
typedef __attribute__((ext_vector_type(4)))  __bf16 v4bf;
typedef __attribute__((ext_vector_type(8)))  float  v8f;

// ---------------- WMMA helpers (CDNA5 layouts, wave32) ----------------

__device__ __forceinline__ v8f wmma_bf16(v16bf a, v16bf b, v8f c) {
  // D = A(16x32 bf16) * B(32x16 bf16) + C(16x16 f32)
  return __builtin_amdgcn_wmma_f32_16x16x32_bf16(
      false, a, false, b, (short)0, c, false, false);
}

// A-matrix fragment, 16x32 bf16. Lane row m = lane&15, half h = lane>>4.
// Per-lane K offsets: {8h..8h+7} and {16+8h..16+8h+7}. p -> row base + k0.
// Both runs are 16-byte aligned -> two b128 loads.
__device__ __forceinline__ v16bf a_frag_bf16(const __bf16* p, int h) {
  const v8bf lo = *(const v8bf*)(p + 8 * h);
  const v8bf hi = *(const v8bf*)(p + 16 + 8 * h);
  return __builtin_shufflevector(lo, hi, 0, 1, 2, 3, 4, 5, 6, 7,
                                 8, 9, 10, 11, 12, 13, 14, 15);
}

// B-matrix fragment, 32x16 bf16, B[k][n]. Lane col n = lane&15, half h = lane>>4.
// Per-lane K offsets: {8h..8h+7} and {16+8h..16+8h+7} -> two b128 loads.
__device__ __forceinline__ v16bf b_frag_bf16(const __bf16* p, int h) {
  const v8bf lo = *(const v8bf*)(p + 8 * h);
  const v8bf hi = *(const v8bf*)(p + 16 + 8 * h);
  return __builtin_shufflevector(lo, hi, 0, 1, 2, 3, 4, 5, 6, 7,
                                 8, 9, 10, 11, 12, 13, 14, 15);
}

// bf16 bit helper (round-to-nearest-even), used for LDS packing.
__device__ __forceinline__ unsigned int f32_to_bf16_bits(float f) {
  unsigned int u = __float_as_uint(f);
  return (u + 0x7FFFu + ((u >> 16) & 1u)) >> 16;
}

// ---------------- Kernel 0: f32 -> bf16 bulk convert -------------------------
__global__ __launch_bounds__(256) void cvt_kernel(
    const float* __restrict__ src, __bf16* __restrict__ dst, int n4) {
  const int i = blockIdx.x * 256 + threadIdx.x;
  if (i < n4) {
    const float4 v = ((const float4*)src)[i];
    v4bf o;
    o[0] = (__bf16)v.x; o[1] = (__bf16)v.y;
    o[2] = (__bf16)v.z; o[3] = (__bf16)v.w;
    ((v4bf*)dst)[i] = o;
  }
}

// ---------------- Kernel 1: projection  Out = (X @ W^T + b) * scale ----------
// X: (8192, 1024) bf16 row-major.  W: (1024, 1024) bf16 row-major.
// vtrans==0: Out in (B, H, L, DPH).  vtrans==1: Out in (B, H, DPH, L) (for V).
__global__ __launch_bounds__(32) void proj_kernel(
    const __bf16* __restrict__ X, const __bf16* __restrict__ W,
    const float* __restrict__ bias, __bf16* __restrict__ Out,
    float scale, int vtrans) {
  const int lane = threadIdx.x;
  const int nl = lane & 15, hh = lane >> 4;
  const int m0 = blockIdx.x * 16;
  const int head = blockIdx.y;
  const int n0 = head * 64;

  v8f c[4] = {};
  const __bf16* arow  = X + (size_t)(m0 + nl) * DD;        // A lane row
  const __bf16* wbase = W + (size_t)(n0 + nl) * DD;        // B lane base row
#pragma unroll 1
  for (int k0 = 0; k0 < DD; k0 += 32) {
    v16bf a = a_frag_bf16(arow + k0, hh);
#pragma unroll
    for (int t = 0; t < 4; ++t) {
      // B[k][n] = W[n][k]; row offset t*16*DD is a compile-time constant.
      c[t] = wmma_bf16(a, b_frag_bf16(wbase + t * 16 * DD + k0, hh), c[t]);
    }
  }
#pragma unroll
  for (int t = 0; t < 4; ++t) {
    const int d = 16 * t + nl;  // col within head
    const float bv = bias[n0 + d];
#pragma unroll
    for (int r = 0; r < 8; ++r) {
      const int mm = m0 + r + 8 * hh;  // global row = b*L + l
      const int bi = mm >> 10, l = mm & (LL - 1);
      const float v = (c[t][r] + bv) * scale;
      const size_t bh = (size_t)(bi * HH_HEADS + head);
      if (vtrans)
        Out[(bh * DPH + d) * LL + l] = (__bf16)v;
      else
        Out[(bh * LL + l) * DPH + d] = (__bf16)v;
    }
  }
}

// ---------------- Kernel 2: qEr[b,h,q,r] = dot(q_scaled[b,h,q,:], rel_emb[r,:])
__global__ __launch_bounds__(64) void qer_kernel(
    const __bf16* __restrict__ Q, const float* __restrict__ rel,
    float* __restrict__ qEr) {
  __shared__ float qrow[DPH];
  const size_t row = blockIdx.x;  // (b*H + h)*L + q
  const int t = threadIdx.x;
  qrow[t] = (float)Q[row * DPH + t];
  __syncthreads();
  for (int r = t; r < NREL; r += 64) {
    float acc = 0.f;
#pragma unroll
    for (int d = 0; d < DPH; ++d) acc += qrow[d] * rel[r * DPH + d];
    qEr[row * NREL + r] = acc;
  }
}

// ---------------- Kernel 3: attention core -----------------------------------
// One wave handles 16 query rows of one (b, head). LDS: 16x1024 f32 = 64 KB.
__global__ __launch_bounds__(32) void attn_kernel(
    const __bf16* __restrict__ Q, const __bf16* __restrict__ Km,
    const __bf16* __restrict__ Vt,  // (B,H,DPH,L)
    const float* __restrict__ qEr, const unsigned char* __restrict__ mask,
    const float* __restrict__ rel, __bf16* __restrict__ ctx,
    float* __restrict__ top_attn) {
  __shared__ float S[16 * LL];

  const int lane = threadIdx.x;
  const int nl = lane & 15, hh = lane >> 4;
  const int bid = blockIdx.x;
  const int qt = bid & 63;
  const int head = (bid >> 6) & 15;
  const int b = bid >> 10;
  const int q0 = qt * 16;
  const size_t bh = (size_t)(b * HH_HEADS + head);

  // ---- scores = Q K^T (M=16 q-rows, K=64 dph, N=1024 seq in 16-wide tiles)
  const __bf16* Qbase = Q + (bh * LL + q0) * DPH;
  const v16bf aq0 = a_frag_bf16(Qbase + (size_t)nl * DPH + 0, hh);
  const v16bf aq1 = a_frag_bf16(Qbase + (size_t)nl * DPH + 32, hh);
  const __bf16* Kbase = Km + (bh * LL + nl) * DPH;  // B[k][n]=K[n_seq][k_dph]
#pragma unroll 1
  for (int kt = 0; kt < LL / 16; ++kt) {
    v8f c = {};
    const __bf16* Krow = Kbase + (size_t)kt * 16 * DPH;
    c = wmma_bf16(aq0, b_frag_bf16(Krow + 0, hh), c);
    c = wmma_bf16(aq1, b_frag_bf16(Krow + 32, hh), c);
#pragma unroll
    for (int r = 0; r < 8; ++r) S[(r + 8 * hh) * LL + kt * 16 + nl] = c[r];
  }
  __syncthreads();

  // ---- add relative bias, mask, softmax per row (store normalized attn, f32)
  const unsigned char* mrow = mask + (size_t)b * LL;
#pragma unroll 1
  for (int m = 0; m < 16; ++m) {
    const int qg = q0 + m;
    const float* qe = qEr + (bh * LL + qg) * NREL;
    float* srow = S + m * LL;
    float mx = -3.0e38f;
    for (int k = lane; k < LL; k += 32) {
      int rr = k - qg;
      rr = rr < -32 ? -32 : (rr > 32 ? 32 : rr);
      float s = srow[k] + qe[rr + 32];
      if (mrow[k]) s = -1e18f;
      srow[k] = s;
      mx = fmaxf(mx, s);
    }
#pragma unroll
    for (int off = 16; off > 0; off >>= 1) mx = fmaxf(mx, __shfl_xor(mx, off, 32));
    float sum = 0.f;
    for (int k = lane; k < LL; k += 32) {
      float e = __expf(srow[k] - mx);
      srow[k] = e;
      sum += e;
    }
#pragma unroll
    for (int off = 16; off > 0; off >>= 1) sum += __shfl_xor(sum, off, 32);
    const float inv = 1.f / sum;
    for (int k = lane; k < LL; k += 32) srow[k] *= inv;
  }
  __syncthreads();

  // ---- top_attn output (head 0 only)
  if (head == 0) {
#pragma unroll 1
    for (int m = 0; m < 16; ++m) {
      const float* srow = S + m * LL;
      float* dst = top_attn + ((size_t)b * LL + q0 + m) * LL;
      for (int k = lane; k < LL; k += 32) dst[k] = srow[k];
    }
  }

  // ---- attn·Er term via bucket decomposition (all reads in f32, pre-pack)
  // bucket 0:  k <= qg-32 ; bucket 64: k >= qg+32 ; buckets 1..63 singletons.
  float slow[16], shigh[16];
#pragma unroll 1
  for (int m = 0; m < 16; ++m) {
    const int qg = q0 + m;
    const float* srow = S + m * LL;
    float sl = 0.f, sh = 0.f;
    for (int k = lane; k <= qg - 32; k += 32) sl += srow[k];
    for (int k = qg + 32 + lane; k < LL; k += 32) sh += srow[k];
#pragma unroll
    for (int off = 16; off > 0; off >>= 1) {
      sl += __shfl_xor(sl, off, 32);
      sh += __shfl_xor(sh, off, 32);
    }
    slow[m] = sl;
    shigh[m] = sh;
  }

  float acc2[4][8];
#pragma unroll
  for (int t = 0; t < 4; ++t) {
    const int d = 16 * t + nl;
#pragma unroll
    for (int r = 0; r < 8; ++r) {
      const int m = r + 8 * hh;
      const int qg = q0 + m;
      float a = slow[m] * rel[d] + shigh[m] * rel[64 * DPH + d];
#pragma unroll 1
      for (int rr = 1; rr < 64; ++rr) {
        const int k = qg + rr - 32;
        if ((unsigned)k < (unsigned)LL) a += S[m * LL + k] * rel[rr * DPH + d];
      }
      acc2[t][r] = a;
    }
  }
  __syncthreads();

  // ---- pack normalized attn f32 -> bf16 in place (row-wise; every store index
  // is strictly below all later load indices, so safe within one wave).
#pragma unroll 1
  for (int m = 0; m < 16; ++m) {
    float* srow = S + m * LL;
    for (int i = lane; i < LL / 2; i += 32) {
      const float lo = srow[2 * i], hi = srow[2 * i + 1];
      const unsigned int w = (f32_to_bf16_bits(hi) << 16) | f32_to_bf16_bits(lo);
      srow[i] = __uint_as_float(w);
    }
  }
  __syncthreads();

  // ---- ctx1 = attn @ V via WMMA (M=16, N=64 in 4 tiles, K=1024)
  // Packed attn row m lives at bf16 index m*2048 + k.
  const __bf16* Sb = (const __bf16*)S;
  v8f c[4] = {};
  const __bf16* Vb = Vt + (bh * DPH + nl) * LL;  // rows = d, contiguous over k(seq)
#pragma unroll 1
  for (int kb = 0; kb < LL; kb += 32) {
    v16bf a = a_frag_bf16(Sb + (size_t)nl * 2048 + kb, hh);
#pragma unroll
    for (int t = 0; t < 4; ++t) {
      // B[k][n] = Vt[n_d][k]; row offset t*16*LL is a compile-time constant.
      c[t] = wmma_bf16(a, b_frag_bf16(Vb + t * 16 * LL + kb, hh), c[t]);
    }
  }

  // ---- store ctx (bf16) in (B, L, H*DPH) so out-projection is a plain GEMM
#pragma unroll
  for (int t = 0; t < 4; ++t) {
    const int d = 16 * t + nl;
#pragma unroll
    for (int r = 0; r < 8; ++r) {
      const int m = r + 8 * hh;
      const float v = c[t][r] + acc2[t][r];
      ctx[((size_t)b * LL + q0 + m) * DD + head * DPH + d] = (__bf16)v;
    }
  }
}

// ---------------- Kernel 4: out = ctx @ Wo^T + bo (f32 result) ---------------
__global__ __launch_bounds__(32) void out_kernel(
    const __bf16* __restrict__ Xc, const __bf16* __restrict__ W,
    const float* __restrict__ bias, float* __restrict__ Out) {
  const int lane = threadIdx.x;
  const int nl = lane & 15, hh = lane >> 4;
  const int m0 = blockIdx.x * 16;
  const int n0 = blockIdx.y * 64;

  v8f c[4] = {};
  const __bf16* arow  = Xc + (size_t)(m0 + nl) * DD;
  const __bf16* wbase = W + (size_t)(n0 + nl) * DD;
#pragma unroll 1
  for (int k0 = 0; k0 < DD; k0 += 32) {
    v16bf a = a_frag_bf16(arow + k0, hh);
#pragma unroll
    for (int t = 0; t < 4; ++t) {
      c[t] = wmma_bf16(a, b_frag_bf16(wbase + t * 16 * DD + k0, hh), c[t]);
    }
  }
#pragma unroll
  for (int t = 0; t < 4; ++t) {
    const int n = n0 + 16 * t + nl;
    const float bv = bias[n];
#pragma unroll
    for (int r = 0; r < 8; ++r) {
      const int mm = m0 + r + 8 * hh;
      Out[(size_t)mm * DD + n] = c[t][r] + bv;
    }
  }
}

// ---------------- Host launcher ----------------------------------------------
extern "C" void kernel_launch(void* const* d_in, const int* in_sizes, int n_in,
                              void* d_out, int out_size, void* d_ws, size_t ws_size,
                              hipStream_t stream) {
  const float* key   = (const float*)d_in[0];
  const float* value = (const float*)d_in[1];
  const float* query = (const float*)d_in[2];
  const unsigned char* mask = (const unsigned char*)d_in[3];  // jnp bool -> 1 byte
  const float* Wq = (const float*)d_in[4];
  const float* bq = (const float*)d_in[5];
  const float* Wk = (const float*)d_in[6];
  const float* bk = (const float*)d_in[7];
  const float* Wv = (const float*)d_in[8];
  const float* bv = (const float*)d_in[9];
  const float* Wo = (const float*)d_in[10];
  const float* bo = (const float*)d_in[11];
  const float* rel = (const float*)d_in[12];

  float* out = (float*)d_out;               // (B, L, D)
  float* top = out + (size_t)BB * LL * DD;  // (B, L, L)

  const size_t MB = (size_t)1 << 20;
  char* ws = (char*)d_ws;
  __bf16* Xq  = (__bf16*)(ws + 0 * MB);    // 16 MiB each
  __bf16* Xk  = (__bf16*)(ws + 16 * MB);
  __bf16* Xv  = (__bf16*)(ws + 32 * MB);
  __bf16* Wqb = (__bf16*)(ws + 48 * MB);   // 2 MiB each
  __bf16* Wkb = (__bf16*)(ws + 50 * MB);
  __bf16* Wvb = (__bf16*)(ws + 52 * MB);
  __bf16* Wob = (__bf16*)(ws + 54 * MB);
  __bf16* Qb  = (__bf16*)(ws + 56 * MB);   // 16 MiB each
  __bf16* Kb  = (__bf16*)(ws + 72 * MB);
  __bf16* Vtb = (__bf16*)(ws + 88 * MB);   // (B,H,DPH,L)
  float*  qEr = (float*)(ws + 104 * MB);   // 34,078,720 B
  __bf16* ctx = (__bf16*)(ws + 104 * MB + 34078720ull);  // 16 MiB

  const int nX4 = BB * LL * DD / 4;   // 2M float4
  const int nW4 = DD * DD / 4;        // 256K float4
  cvt_kernel<<<dim3(nX4 / 256), dim3(256), 0, stream>>>(query, Xq, nX4);
  cvt_kernel<<<dim3(nX4 / 256), dim3(256), 0, stream>>>(key,   Xk, nX4);
  cvt_kernel<<<dim3(nX4 / 256), dim3(256), 0, stream>>>(value, Xv, nX4);
  cvt_kernel<<<dim3(nW4 / 256), dim3(256), 0, stream>>>(Wq, Wqb, nW4);
  cvt_kernel<<<dim3(nW4 / 256), dim3(256), 0, stream>>>(Wk, Wkb, nW4);
  cvt_kernel<<<dim3(nW4 / 256), dim3(256), 0, stream>>>(Wv, Wvb, nW4);
  cvt_kernel<<<dim3(nW4 / 256), dim3(256), 0, stream>>>(Wo, Wob, nW4);

  const dim3 gproj(BB * LL / 16, HH_HEADS);  // 512 x 16
  const dim3 wave(32);
  proj_kernel<<<gproj, wave, 0, stream>>>(Xq, Wqb, bq, Qb, 0.125f, 0);  // 1/sqrt(64)
  proj_kernel<<<gproj, wave, 0, stream>>>(Xk, Wkb, bk, Kb, 1.0f, 0);
  proj_kernel<<<gproj, wave, 0, stream>>>(Xv, Wvb, bv, Vtb, 1.0f, 1);

  qer_kernel<<<dim3(BB * HH_HEADS * LL), dim3(64), 0, stream>>>(Qb, rel, qEr);

  attn_kernel<<<dim3(BB * HH_HEADS * (LL / 16)), wave, 0, stream>>>(
      Qb, Kb, Vtb, qEr, mask, rel, ctx, top);

  out_kernel<<<gproj, wave, 0, stream>>>(ctx, Wob, bo, out);
}
